// SelfAttention_10909216932535
// MI455X (gfx1250) — matrix operand order repaired and checked by reading, not compiled, not run on hardware
//
#include <hip/hip_runtime.h>
#include <hip/hip_bf16.h>

// ---------------------------------------------------------------------------
// Self-attention block for MI455X (gfx1250): wave32, WMMA f16 (f32 accum),
// TDM (tensor_load_to_lds) double-buffered LDS pipelines for the GEMMs and
// for flash-attention K tiles.
//   B=32, C=512, H=W=32 -> N=1024 tokens, O3=3C=1536, 32 groups GN.
// Pipeline: GN -> QKV GEMM -> flash attention -> proj GEMM (+bias+residual).
// ---------------------------------------------------------------------------

typedef __attribute__((ext_vector_type(16))) _Float16 v16h;
typedef __attribute__((ext_vector_type(8)))  _Float16 v8h;
typedef __attribute__((ext_vector_type(8)))  float    v8f;
typedef __attribute__((ext_vector_type(4)))  unsigned int v4ui;
typedef __attribute__((ext_vector_type(8)))  int      v8i;
typedef __attribute__((ext_vector_type(4)))  int      v4i;

union V16H { v16h v; struct { v8h lo, hi; } p; };

#define BATCH 32
#define CDIM  512
#define NTOK  1024
#define ODIM  1536
#define GROUPS 32
#define KROW  520   // LDS row stride (f16) for staged K tile: 512 + 8 pad
#define KSTEP 64    // GEMM staged k-step
#define TROW  72    // LDS row stride (f16) for staged GEMM tiles: 64 + 8 pad

// TDM only in the device pass, and only if the toolchain declares the builtin.
#if defined(__HIP_DEVICE_COMPILE__) && defined(__has_builtin)
#if __has_builtin(__builtin_amdgcn_tensor_load_to_lds)
#define USE_TDM 1
#endif
#endif
#ifndef USE_TDM
#define USE_TDM 0
#endif

// ---------------- fragment loaders (contiguous k = 16h + s, same order for A and B) --------------------
// A fragment (16x32 f16, row-major, lda elems): lane l: row=(l&15), h=(l>>4),
// elems 0..15 = K h*16..+15 contiguous
__device__ __forceinline__ v16h load_a_frag(const _Float16* A, int lda, int lane) {
  const int r = lane & 15, h = lane >> 4;
  const _Float16* p = A + (size_t)r * lda + h * 16;
  V16H u;
  u.p.lo = *(const v8h*)(p);
  u.p.hi = *(const v8h*)(p + 8);
  return u.v;
}

// B fragment (32x16) from B^T row-major (N x K): lane l holds col n=(l&15),
// elems = K (l>>4)*16..+15 contiguous.
__device__ __forceinline__ v16h load_bT_frag(const _Float16* BT, int ldb, int lane) {
  const int c = lane & 15, h = lane >> 4;
  return *(const v16h*)(BT + (size_t)c * ldb + h * 16);
}

// LDS variants with padded row strides (16B-aligned halves).
__device__ __forceinline__ v16h load_a_lds(const _Float16* As, int lane) {
  const int r = lane & 15, h = lane >> 4;
  const _Float16* p = As + r * TROW + h * 16;
  V16H u;
  u.p.lo = *(const v8h*)(p);
  u.p.hi = *(const v8h*)(p + 8);
  return u.v;
}
__device__ __forceinline__ v16h load_w_lds(const _Float16* Ws, int lane) {
  const int c = lane & 15, h = lane >> 4;
  const _Float16* p = Ws + c * TROW + h * 16;
  V16H u;
  u.p.lo = *(const v8h*)(p);
  u.p.hi = *(const v8h*)(p + 8);
  return u.v;
}
__device__ __forceinline__ v16h load_k_lds(const _Float16* base, int row0, int kf, int lane) {
  const int c = lane & 15, h = lane >> 4;
  const _Float16* p = base + (size_t)(row0 + c) * KROW + kf * 32 + h * 16;
  V16H u;
  u.p.lo = *(const v8h*)(p);
  u.p.hi = *(const v8h*)(p + 8);
  return u.v;
}

__device__ __forceinline__ v8f wmma_f16(v16h a, v16h b, v8f c) {
  return __builtin_amdgcn_wmma_f32_16x16x32_f16(false, a, false, b, (short)0, c,
                                                false, false);
}

__device__ __forceinline__ unsigned lds_off(const void* p) {
  // Flat LDS aperture addresses carry the LDS byte offset in addr[31:0].
  return (unsigned)(uintptr_t)p;
}

#if USE_TDM
// TDM 2-D tile load: w elems per row (data_size=2B), hgt rows, global row
// stride `stride` elems, LDS destination lds_byte with (pad_amt+1) DWORDs of
// padding inserted every 2^(pad_int+1) DWORDs stored.
// D# packing per CDNA5 ISA ch.8 (group0: count/lds/global/type; group1:
// data_size, pad, dims, strides; groups 2/3 zero for 2-D tiles).
__device__ __forceinline__ void tdm_load_2d(const _Float16* gptr, unsigned lds_byte,
                                            unsigned w, unsigned hgt, unsigned stride,
                                            unsigned pad_int, unsigned pad_amt) {
  const unsigned long long ga = (unsigned long long)(uintptr_t)gptr;
  v4ui g0;
  g0[0] = 1u;                                    // count=1 valid descriptor
  g0[1] = lds_byte;                              // lds_addr (bytes)
  g0[2] = (unsigned)(ga & 0xFFFFFFFFu);          // global_addr[31:0]
  g0[3] = (unsigned)((ga >> 32) & 0x1FFFFFFu) | (2u << 30);   // addr[56:32], type=2
  v8i g1;
  g1[0] = (int)((1u << 16) | (1u << 20) | (pad_int << 22) | (pad_amt << 25));
  g1[1] = (int)((w & 0xFFFFu) << 16);            // tensor_dim0 lo
  g1[2] = (int)((w >> 16) | ((hgt & 0xFFFFu) << 16));   // dim0 hi | tensor_dim1 lo
  g1[3] = (int)((hgt >> 16) | ((w & 0xFFFFu) << 16));   // dim1 hi | tile_dim0
  g1[4] = (int)(hgt & 0xFFFFu);                  // tile_dim1 | tile_dim2=0
  g1[5] = (int)stride;                           // tensor_dim0_stride lo32
  g1[6] = 0;
  g1[7] = 0;
  const v4i z4 = {0, 0, 0, 0};
#if __clang_major__ >= 23
  const v8i z8 = {0, 0, 0, 0, 0, 0, 0, 0};
  __builtin_amdgcn_tensor_load_to_lds(g0, g1, z4, z4, z8, 0);
#else
  __builtin_amdgcn_tensor_load_to_lds(g0, g1, z4, z4, 0);
#endif
}
#endif  // USE_TDM

// ---------------------------------------------------------------------------
// Shared GEMM main loop: block tile 128(M) x 64(N), K=512 in 8 steps of 64.
// TDM double-buffers A (128x64) and W (64x64) tiles in LDS with counted
// tensorcnt waits. Per stage: load 2 A-frags + 8 W-frags first, then 8
// back-to-back WMMAs (partial dscnt waits, fewer barriers per WMMA).
// ---------------------------------------------------------------------------
__device__ __forceinline__ void gemm_mainloop(const _Float16* Ag, const _Float16* Wg,
                                              _Float16* a0, _Float16* a1,
                                              _Float16* w0, _Float16* w1p,
                                              int wave, int lane, v8f acc[4]) {
#if USE_TDM
  const unsigned ao0 = lds_off(a0), ao1 = lds_off(a1);
  const unsigned wo0 = lds_off(w0), wo1 = lds_off(w1p);
  if (wave == 0) {
    tdm_load_2d(Ag, ao0, KSTEP, 128, CDIM, 4, 3);   // pad 16B per 128B row
    tdm_load_2d(Wg, wo0, KSTEP, 64, CDIM, 4, 3);
  }
  #pragma unroll 1
  for (int s = 0; s < CDIM / KSTEP; ++s) {
    if (wave == 0) {
      if (s + 1 < CDIM / KSTEP) {   // prefetch next tiles, wait only for current
        tdm_load_2d(Ag + (s + 1) * KSTEP, ((s + 1) & 1) ? ao1 : ao0, KSTEP, 128, CDIM, 4, 3);
        tdm_load_2d(Wg + (s + 1) * KSTEP, ((s + 1) & 1) ? wo1 : wo0, KSTEP, 64, CDIM, 4, 3);
        __builtin_amdgcn_s_wait_tensorcnt(2);
      } else {
        __builtin_amdgcn_s_wait_tensorcnt(0);
      }
    }
    __syncthreads();          // current tiles visible to all waves
    const _Float16* As = ((s & 1) ? a1 : a0) + wave * 16 * TROW;
    const _Float16* Ws = (s & 1) ? w1p : w0;
    v16h af[2], wf[8];
    af[0] = load_a_lds(As, lane);
    af[1] = load_a_lds(As + 32, lane);
    #pragma unroll
    for (int t = 0; t < 4; ++t) {
      wf[2 * t]     = load_w_lds(Ws + t * 16 * TROW, lane);
      wf[2 * t + 1] = load_w_lds(Ws + t * 16 * TROW + 32, lane);
    }
    #pragma unroll
    for (int t = 0; t < 4; ++t) {
      acc[t] = wmma_f16(af[0], wf[2 * t], acc[t]);
      acc[t] = wmma_f16(af[1], wf[2 * t + 1], acc[t]);
    }
    __syncthreads();          // buffer free before step s+2 overwrites it
  }
#else
  (void)a1; (void)w1p;
  #pragma unroll 1
  for (int s = 0; s < CDIM / KSTEP; ++s) {
    __syncthreads();
    for (int idx = threadIdx.x; idx < 128 * 8; idx += 256) {
      const int row = idx >> 3, ch = idx & 7;
      *(v8h*)(a0 + row * TROW + ch * 8) =
          *(const v8h*)(Ag + (size_t)row * CDIM + s * KSTEP + ch * 8);
    }
    for (int idx = threadIdx.x; idx < 64 * 8; idx += 256) {
      const int row = idx >> 3, ch = idx & 7;
      *(v8h*)(w0 + row * TROW + ch * 8) =
          *(const v8h*)(Wg + (size_t)row * CDIM + s * KSTEP + ch * 8);
    }
    __syncthreads();
    const _Float16* As = a0 + wave * 16 * TROW;
    v16h af[2], wf[8];
    af[0] = load_a_lds(As, lane);
    af[1] = load_a_lds(As + 32, lane);
    #pragma unroll
    for (int t = 0; t < 4; ++t) {
      wf[2 * t]     = load_w_lds(w0 + t * 16 * TROW, lane);
      wf[2 * t + 1] = load_w_lds(w0 + t * 16 * TROW + 32, lane);
    }
    #pragma unroll
    for (int t = 0; t < 4; ++t) {
      acc[t] = wmma_f16(af[0], wf[2 * t], acc[t]);
      acc[t] = wmma_f16(af[1], wf[2 * t + 1], acc[t]);
    }
  }
#endif
}

// ---------------------------------------------------------------------------
// Kernel 1: GroupNorm -> token-major f16 xn [B][N][C]
// ---------------------------------------------------------------------------
__global__ __launch_bounds__(256) void gn_kernel(const float* __restrict__ x,
                                                 const float* __restrict__ gscale,
                                                 const float* __restrict__ gbias,
                                                 _Float16* __restrict__ xh) {
  const int b = blockIdx.x >> 5;
  const int g = blockIdx.x & 31;
  const float* xp = x + ((size_t)b * CDIM + g * 16) * NTOK;

  float s = 0.f, s2 = 0.f;
  for (int idx = threadIdx.x; idx < 16 * NTOK; idx += 256) {
    float v = xp[idx];
    s += v;
    s2 += v * v;
  }
  #pragma unroll
  for (int off = 16; off > 0; off >>= 1) {
    s  += __shfl_xor(s, off, 32);
    s2 += __shfl_xor(s2, off, 32);
  }
  __shared__ float rbuf[16];
  const int wave = threadIdx.x >> 5, lane = threadIdx.x & 31;
  if (lane == 0) { rbuf[wave] = s; rbuf[8 + wave] = s2; }
  __syncthreads();
  if (threadIdx.x == 0) {
    float ts = 0.f, ts2 = 0.f;
    #pragma unroll
    for (int w = 0; w < 8; ++w) { ts += rbuf[w]; ts2 += rbuf[8 + w]; }
    const float inv = 1.0f / (16.0f * NTOK);
    float mean = ts * inv;
    float var = ts2 * inv - mean * mean;
    rbuf[0] = mean;
    rbuf[1] = rsqrtf(var + 1e-5f);
  }
  __syncthreads();
  const float mean = rbuf[0], rstd = rbuf[1];

  for (int idx = threadIdx.x; idx < 16 * NTOK; idx += 256) {
    const int c = idx >> 10;
    const int n = idx & (NTOK - 1);
    const int cg = g * 16 + c;
    float y = (xp[idx] - mean) * rstd * gscale[cg] + gbias[cg];
    xh[((size_t)b * NTOK + n) * CDIM + cg] = (_Float16)y;
  }
}

// ---------------------------------------------------------------------------
// Kernel 2: f32 -> f16 weight conversion
// ---------------------------------------------------------------------------
__global__ __launch_bounds__(256) void wconv_kernel(const float* __restrict__ w1,
                                                    const float* __restrict__ w2,
                                                    _Float16* __restrict__ w1h,
                                                    _Float16* __restrict__ w2h) {
  const int i = blockIdx.x * 256 + threadIdx.x;
  if (i < ODIM * CDIM) w1h[i] = (_Float16)w1[i];
  if (i < CDIM * CDIM) w2h[i] = (_Float16)w2[i];
}

// ---------------------------------------------------------------------------
// Kernel 3: QKV GEMM (TDM-pipelined LDS main loop).
// q pre-scaled by C^-0.5; v also stored transposed to vT [B][C][N].
// ---------------------------------------------------------------------------
__global__ __launch_bounds__(256) void qkv_gemm_kernel(const _Float16* __restrict__ xh,
                                                       const _Float16* __restrict__ w1h,
                                                       const float* __restrict__ b1,
                                                       _Float16* __restrict__ qkvh,
                                                       _Float16* __restrict__ vT) {
  __shared__ __align__(64) _Float16 asmem[2][128 * TROW];
  __shared__ __align__(64) _Float16 wsmem[2][64 * TROW];
  const int b = blockIdx.z;
  const int wave = threadIdx.x >> 5, lane = threadIdx.x & 31;
  const int nb = blockIdx.y * 128;
  const int n0 = nb + wave * 16;
  const int o0 = blockIdx.x * 64;
  const _Float16* Ag = xh + ((size_t)b * NTOK + nb) * CDIM;
  const _Float16* Wg = w1h + (size_t)o0 * CDIM;

  v8f acc[4] = {};
  gemm_mainloop(Ag, Wg, asmem[0], asmem[1], wsmem[0], wsmem[1], wave, lane, acc);

  const float qscale = 0.044194173824159216f;  // 512^-0.5
  const int colL = lane & 15, h = lane >> 4;
  #pragma unroll
  for (int t = 0; t < 4; ++t) {
    const int o = o0 + t * 16 + colL;
    const float bias = b1[o];
    const float sc = (o < CDIM) ? qscale : 1.0f;
    v8h pk;
    #pragma unroll
    for (int i = 0; i < 8; ++i) {
      const float v = (acc[t][i] + bias) * sc;
      const int n = n0 + h * 8 + i;
      const _Float16 hv = (_Float16)v;
      qkvh[((size_t)b * NTOK + n) * ODIM + o] = hv;
      pk[i] = hv;
    }
    if (o >= 2 * CDIM) {  // v: rows i are consecutive tokens -> packed store
      const int c = o - 2 * CDIM;
      *(v8h*)(vT + ((size_t)b * CDIM + c) * NTOK + n0 + h * 8) = pk;
    }
  }
}

// ---------------------------------------------------------------------------
// Kernel 4: flash attention, one wave per 16-row Q tile; K tiles staged in
// LDS by the TDM (tile j+1 prefetched during softmax/PV of tile j),
// online softmax via 16-lane shfl reductions, P relaid out D->A through LDS.
// ---------------------------------------------------------------------------
__global__ __launch_bounds__(128) void attn_kernel(const _Float16* __restrict__ qkvh,
                                                   const _Float16* __restrict__ vT,
                                                   _Float16* __restrict__ attnh) {
  __shared__ __align__(64) _Float16 kbuf[32 * KROW];   // 32 x (512+pad) f16
  __shared__ __align__(32) _Float16 pbuf[4][16 * 48];
  const int wave = threadIdx.x >> 5, lane = threadIdx.x & 31;
  const int gwave = blockIdx.x * 4 + wave;
  const int b = gwave >> 6;           // 64 q-tiles per batch, 4 per block (same b)
  const int n0 = (gwave & 63) * 16;
  const int colL = lane & 15, h = lane >> 4;

  const _Float16* qbase = qkvh + ((size_t)b * NTOK + n0) * ODIM;   // q: o in [0,512)
  const _Float16* kbase = qkvh + (size_t)b * NTOK * ODIM + CDIM;   // k rows
  const _Float16* vbase = vT + (size_t)b * CDIM * NTOK;
  _Float16* pb = &pbuf[wave][0];
  const unsigned kbuf_lds = lds_off(&kbuf[0]);

  v16h qf[16];
  #pragma unroll
  for (int kf = 0; kf < 16; ++kf) qf[kf] = load_a_frag(qbase + kf * 32, ODIM, lane);

  v8f acc[32] = {};                   // O tile: 16 rows x 512 cols
  float mrun[8], lrun[8];
  #pragma unroll
  for (int i = 0; i < 8; ++i) { mrun[i] = -1e30f; lrun[i] = 0.f; }

#if USE_TDM
  if (wave == 0)                       // stage K(0): pad 16B per 1KB row
    tdm_load_2d(kbase, kbuf_lds, 512, 32, ODIM, 7, 3);
#else
  (void)kbuf_lds;
#endif

  #pragma unroll 1
  for (int j = 0; j < NTOK / 32; ++j) {
    const int m0 = j * 32;
#if USE_TDM
    if (wave == 0) __builtin_amdgcn_s_wait_tensorcnt(0);
    __syncthreads();                  // K(j) visible to all waves
#else
    __syncthreads();
    for (int idx = threadIdx.x; idx < 32 * 64; idx += 128) {   // cooperative copy
      const int row = idx >> 6, chunk = idx & 63;
      *(v8h*)(kbuf + (size_t)row * KROW + chunk * 8) =
          *(const v8h*)(kbase + ((size_t)(m0 + row)) * ODIM + chunk * 8);
    }
    __syncthreads();
#endif
    // ---- S = Q * K^T for 32 key columns (double-buffered LDS frags) ----
    v8f Sa = {}, Sb = {};
    v16h ka = load_k_lds(kbuf, 0, 0, lane);
    v16h kb = load_k_lds(kbuf, 16, 0, lane);
    #pragma unroll
    for (int kf = 0; kf < 16; ++kf) {
      v16h kan = ka, kbn = kb;
      if (kf < 15) {
        kan = load_k_lds(kbuf, 0, kf + 1, lane);
        kbn = load_k_lds(kbuf, 16, kf + 1, lane);
      }
      Sa = wmma_f16(qf[kf], ka, Sa);
      Sb = wmma_f16(qf[kf], kb, Sb);
      ka = kan;
      kb = kbn;
    }
    __syncthreads();                  // everyone done reading kbuf
#if USE_TDM
    if (wave == 0 && j + 1 < NTOK / 32)   // prefetch K(j+1) during softmax/PV
      tdm_load_2d(kbase + (size_t)(m0 + 32) * ODIM, kbuf_lds, 512, 32, ODIM, 7, 3);
#endif
    // ---- online softmax (rows i+8*h live in one 16-lane half) ----
    float alpha[8];
    #pragma unroll
    for (int i = 0; i < 8; ++i) {
      float mc = fmaxf(Sa[i], Sb[i]);
      mc = fmaxf(mc, __shfl_xor(mc, 1, 32));
      mc = fmaxf(mc, __shfl_xor(mc, 2, 32));
      mc = fmaxf(mc, __shfl_xor(mc, 4, 32));
      mc = fmaxf(mc, __shfl_xor(mc, 8, 32));
      const float mnew = fmaxf(mrun[i], mc);
      alpha[i] = __expf(mrun[i] - mnew);
      const float pa = __expf(Sa[i] - mnew);
      const float pv = __expf(Sb[i] - mnew);
      float rs = pa + pv;
      rs += __shfl_xor(rs, 1, 32);
      rs += __shfl_xor(rs, 2, 32);
      rs += __shfl_xor(rs, 4, 32);
      rs += __shfl_xor(rs, 8, 32);
      lrun[i] = lrun[i] * alpha[i] + rs;
      mrun[i] = mnew;
      pb[(i + 8 * h) * 48 + colL]      = (_Float16)pa;   // D-layout -> LDS
      pb[(i + 8 * h) * 48 + colL + 16] = (_Float16)pv;
    }
    #pragma unroll
    for (int t = 0; t < 32; ++t)
      #pragma unroll
      for (int i = 0; i < 8; ++i) acc[t][i] *= alpha[i];
    __syncthreads();
    // ---- re-read P as A fragment ----
    V16H pu;
    pu.p.lo = *(const v8h*)(pb + colL * 48 + h * 16);
    pu.p.hi = *(const v8h*)(pb + colL * 48 + h * 16 + 8);
    const v16h pf = pu.v;
    __syncthreads();
    // ---- O += P * V (32 channel blocks, double-buffered global frags) ----
    v16h vf = load_bT_frag(vbase + m0, NTOK, lane);
    #pragma unroll
    for (int t = 0; t < 32; ++t) {
      v16h vfn = vf;
      if (t < 31) vfn = load_bT_frag(vbase + (size_t)((t + 1) * 16) * NTOK + m0, NTOK, lane);
      acc[t] = wmma_f16(pf, vf, acc[t]);
      vf = vfn;
    }
  }

  // finalize: divide by softmax denominator, store f16 [B][N][C]
  float linv[8];
  #pragma unroll
  for (int i = 0; i < 8; ++i) linv[i] = 1.0f / lrun[i];
  #pragma unroll
  for (int t = 0; t < 32; ++t)
    #pragma unroll
    for (int i = 0; i < 8; ++i)
      attnh[((size_t)b * NTOK + n0 + 8 * h + i) * CDIM + t * 16 + colL] =
          (_Float16)(acc[t][i] * linv[i]);
}

// ---------------------------------------------------------------------------
// Kernel 5: output projection + bias + residual (TDM-pipelined LDS main loop).
// ---------------------------------------------------------------------------
__global__ __launch_bounds__(256) void proj_gemm_kernel(const _Float16* __restrict__ attnh,
                                                        const _Float16* __restrict__ w2h,
                                                        const float* __restrict__ b2,
                                                        const float* __restrict__ x,
                                                        float* __restrict__ out) {
  __shared__ __align__(64) _Float16 asmem[2][128 * TROW];
  __shared__ __align__(64) _Float16 wsmem[2][64 * TROW];
  const int b = blockIdx.z;
  const int wave = threadIdx.x >> 5, lane = threadIdx.x & 31;
  const int nb = blockIdx.y * 128;
  const int n0 = nb + wave * 16;
  const int o0 = blockIdx.x * 64;
  const _Float16* Ag = attnh + ((size_t)b * NTOK + nb) * CDIM;
  const _Float16* Wg = w2h + (size_t)o0 * CDIM;

  v8f acc[4] = {};
  gemm_mainloop(Ag, Wg, asmem[0], asmem[1], wsmem[0], wsmem[1], wave, lane, acc);

  const int colL = lane & 15, h = lane >> 4;
  #pragma unroll
  for (int t = 0; t < 4; ++t) {
    const int o = o0 + t * 16 + colL;
    const float bias = b2[o];
    const size_t base = ((size_t)b * CDIM + o) * NTOK + n0 + h * 8;
    #pragma unroll
    for (int i = 0; i < 8; ++i) out[base + i] = acc[t][i] + bias + x[base + i];
  }
}

// ---------------------------------------------------------------------------
extern "C" void kernel_launch(void* const* d_in, const int* in_sizes, int n_in,
                              void* d_out, int out_size, void* d_ws, size_t ws_size,
                              hipStream_t stream) {
  (void)in_sizes; (void)n_in; (void)out_size; (void)ws_size;
  const float* x      = (const float*)d_in[0];
  const float* gscale = (const float*)d_in[1];
  const float* gbias  = (const float*)d_in[2];
  const float* w1     = (const float*)d_in[3];
  const float* b1     = (const float*)d_in[4];
  const float* w2     = (const float*)d_in[5];
  const float* b2     = (const float*)d_in[6];
  float* out = (float*)d_out;

  // Workspace layout (bytes), total ~169.9 MB:
  char* ws = (char*)d_ws;
  _Float16* xh   = (_Float16*)(ws);                 // 33,554,432  xn [B][N][C] (reused as attn)
  _Float16* w1h  = (_Float16*)(ws + 33554432);      //  1,572,864  w1 f16
  _Float16* w2h  = (_Float16*)(ws + 35127296);      //    524,288  w2 f16
  _Float16* qkvh = (_Float16*)(ws + 35651584);      // 100,663,296 qkv [B][N][3C]
  _Float16* vT   = (_Float16*)(ws + 136314880);     // 33,554,432  v^T [B][C][N]
  _Float16* attnh = xh;                             // reuse (xn dead after QKV GEMM)

  gn_kernel<<<BATCH * GROUPS, 256, 0, stream>>>(x, gscale, gbias, xh);
  wconv_kernel<<<(ODIM * CDIM + 255) / 256, 256, 0, stream>>>(w1, w2, w1h, w2h);
  qkv_gemm_kernel<<<dim3(ODIM / 64, NTOK / 128, BATCH), 256, 0, stream>>>(xh, w1h, b1, qkvh, vT);
  attn_kernel<<<(BATCH * (NTOK / 16)) / 4, 128, 0, stream>>>(qkvh, vT, attnh);
  proj_gemm_kernel<<<dim3(CDIM / 64, NTOK / 128, BATCH), 256, 0, stream>>>(attnh, w2h, b2, x, out);
}
